// MoELayer_90933047591233
// MI455X (gfx1250) — matrix operand order repaired and checked by reading, not compiled, not run on hardware
//
#include <hip/hip_runtime.h>
#include <hip/hip_bf16.h>

// ---------------------------------------------------------------------------
// Problem constants (from reference): B=4, S=2048 -> T=8192, D=1024, E=8, F=512
// ---------------------------------------------------------------------------
static constexpr int kT = 8192;   // tokens
static constexpr int kD = 1024;   // hidden dim
static constexpr int kE = 8;      // experts
static constexpr int kF = 512;    // ffn dim
static constexpr int kTopK = 2;

// GEMM tile config (double-buffered async/TDM pipeline)
static constexpr int BM = 32;     // token rows per block
static constexpr int BN = 64;     // output cols per chunk (8 waves x 16x16 tiles)
static constexpr int BK = 32;     // K per WMMA step (bf16 16x16x32)
static constexpr int PAD = 8;     // LDS row padding ((32+8)*2 = 80B stride, 16B aligned)
static constexpr int HS_STRIDE = kF + 8;  // 520 elems -> 1040B row stride (16B multiple)

typedef __bf16 bf16_t;
typedef bf16_t v16bf __attribute__((ext_vector_type(16)));
typedef float  v8f   __attribute__((ext_vector_type(8)));
typedef unsigned v4u __attribute__((ext_vector_type(4)));
typedef unsigned v8u __attribute__((ext_vector_type(8)));

union Frag { v16bf v; uint4 q[2]; };

__device__ __forceinline__ unsigned short f2bf(float f) {
  unsigned int u = __float_as_uint(f);
  u += 0x7FFFu + ((u >> 16) & 1u);   // round-to-nearest-even
  return (unsigned short)(u >> 16);
}

// CDNA5 async global->LDS copy (ASYNCcnt-tracked, per-lane gather).
__device__ __forceinline__ void async_b64(unsigned lds_off, const void* gaddr) {
  asm volatile("global_load_async_to_lds_b64 %0, %1, off"
               :: "v"(lds_off), "v"(gaddr) : "memory");
}
__device__ __forceinline__ void wait_async0() {
  asm volatile("s_wait_asynccnt 0" ::: "memory");
}
__device__ __forceinline__ unsigned lds_off(const void* p) {
  // flat LDS addresses map to LDS via addr[31:0] (ISA aperture rule)
  return (unsigned)(uintptr_t)p;
}

// CDNA5 Tensor Data Mover: one-instruction 2D tile DMA global->LDS (TENSORcnt).
// D# group0 = {count=1, lds_addr, global_addr[56:0], type=2}; group1 = tensor/tile
// geometry (built once as constants). LDS row pad handled by pad_interval/amount.
__device__ __forceinline__ void tdm_load_2d(unsigned lds_base, const void* gaddr,
                                            v8u g1) {
  unsigned long long ga = (unsigned long long)(uintptr_t)gaddr;
  v4u g0;
  g0.x = 1u;                                   // count=1, no gather, user descriptor
  g0.y = lds_base;                             // LDS byte address of tile dest
  g0.z = (unsigned)ga;                         // global_addr[31:0]
  g0.w = ((unsigned)(ga >> 32) & 0x01FFFFFFu)  // global_addr[56:32]
         | 0x80000000u;                        // type=2 ("image")
  asm volatile("tensor_load_to_lds %0, %1" :: "s"(g0), "s"(g1) : "memory");
}

// group1 geometry: data_size=1(2B), pad_enable, pad_interval=3(64B), pad_amount=3(16B),
// tile_dim0=BK(32 elems), tile_dim1=BN(64 rows); per-phase tensor dim0/dim1/stride.
__device__ __forceinline__ v8u tdm_group1(unsigned dim0, unsigned dim1, unsigned stride) {
  v8u g;
  g[0] = (1u << 16) | (1u << 20) | (3u << 22) | (3u << 25);   // 0x06D10000
  g[1] = (dim0 & 0xFFFFu) << 16;                // tensor_dim0[15:0] at bits 63:48
  g[2] = ((dim0 >> 16) & 0xFFFFu) | ((dim1 & 0xFFFFu) << 16); // dim0 hi | dim1 lo
  g[3] = ((dim1 >> 16) & 0xFFFFu) | ((unsigned)BK << 16);     // dim1 hi | tile_dim0
  g[4] = (unsigned)BN;                          // tile_dim1 (tile_dim2=0)
  g[5] = stride;                                // tensor_dim0_stride[31:0]
  g[6] = 0u;                                    // stride hi | dim1_stride lo
  g[7] = 0u;
  return g;
}

// ---------------------------------------------------------------------------
// f32 -> bf16 bulk convert (vectorized, grid-stride)
// ---------------------------------------------------------------------------
__global__ __launch_bounds__(256) void moe_cvt_bf16(const float* __restrict__ s,
                                                    unsigned short* __restrict__ d, int n) {
  int i = blockIdx.x * blockDim.x + threadIdx.x;
  int stride = gridDim.x * blockDim.x;
  for (int j = i * 4; j < n; j += stride * 4) {
    float4 v = *(const float4*)(s + j);
    uint2 p;
    p.x = (unsigned)f2bf(v.x) | ((unsigned)f2bf(v.y) << 16);
    p.y = (unsigned)f2bf(v.z) | ((unsigned)f2bf(v.w) << 16);
    *(uint2*)(d + j) = p;
  }
}

// ---------------------------------------------------------------------------
// Tiled transpose + f32->bf16 convert: src [b][R][C] f32 -> dst [b][C][R] bf16.
// Puts weight panels contiguous along K so GEMM staging is pure block copies.
// ---------------------------------------------------------------------------
__global__ __launch_bounds__(256) void moe_cvt_t_bf16(const float* __restrict__ src,
                                                      unsigned short* __restrict__ dst,
                                                      int R, int C) {
  __shared__ unsigned short tile[32][33];
  const int b  = blockIdx.z;
  const int r0 = blockIdx.y * 32;
  const int c0 = blockIdx.x * 32;
  const int tx = threadIdx.x;        // 0..31
  const int ty = threadIdx.y;        // 0..7
  const float* S = src + (size_t)b * R * C;
  unsigned short* Dm = dst + (size_t)b * C * R;
#pragma unroll
  for (int i = 0; i < 4; ++i)
    tile[ty + i * 8][tx] = f2bf(S[(size_t)(r0 + ty + i * 8) * C + c0 + tx]);
  __syncthreads();
#pragma unroll
  for (int i = 0; i < 4; ++i)
    Dm[(size_t)(c0 + ty + i * 8) * R + r0 + tx] = tile[tx][ty + i * 8];
}

// ---------------------------------------------------------------------------
// Router: one wave per token. 8 logits, top-2, renormalize (== softmax over top2)
// ---------------------------------------------------------------------------
__global__ __launch_bounds__(256) void moe_router(const float* __restrict__ x,
                                                  const float* __restrict__ rw,
                                                  int* __restrict__ topk_i,
                                                  float* __restrict__ topk_w,
                                                  int* __restrict__ counts) {
  const int lane = threadIdx.x & 31;
  const int wave = threadIdx.x >> 5;
  const int t = blockIdx.x * 8 + wave;
  if (t >= kT) return;
  const float* xr = x + (size_t)t * kD;

  float acc[kE];
#pragma unroll
  for (int e = 0; e < kE; ++e) acc[e] = 0.f;

  for (int d = lane * 4; d < kD; d += 32 * 4) {
    float4 xv = *(const float4*)(xr + d);
#pragma unroll
    for (int e = 0; e < kE; ++e) {
      float4 wv = *(const float4*)(rw + (size_t)e * kD + d);
      acc[e] = fmaf(xv.x, wv.x, fmaf(xv.y, wv.y, fmaf(xv.z, wv.z, fmaf(xv.w, wv.w, acc[e]))));
    }
  }
#pragma unroll
  for (int e = 0; e < kE; ++e)
    for (int off = 16; off > 0; off >>= 1) acc[e] += __shfl_xor(acc[e], off, 32);

  if (lane == 0) {
    int i1 = 0;
    for (int e = 1; e < kE; ++e) if (acc[e] > acc[i1]) i1 = e;
    int i2 = (i1 == 0) ? 1 : 0;
    for (int e = 0; e < kE; ++e) if (e != i1 && acc[e] > acc[i2]) i2 = e;
    // renormalized top-k softmax probs == softmax over the two top logits
    float e2 = __expf(acc[i2] - acc[i1]);
    float inv = 1.f / (1.f + e2);
    topk_i[t * 2 + 0] = i1;  topk_w[t * 2 + 0] = inv;
    topk_i[t * 2 + 1] = i2;  topk_w[t * 2 + 1] = e2 * inv;
    atomicAdd(&counts[i1], 1);
    atomicAdd(&counts[i2], 1);
  }
}

// ---------------------------------------------------------------------------
// Scan (E=8, single thread) and scatter into per-expert token lists
// ---------------------------------------------------------------------------
__global__ void moe_scan(const int* __restrict__ counts, int* __restrict__ offsets,
                         int* __restrict__ cursors) {
  if (threadIdx.x == 0 && blockIdx.x == 0) {
    int s = 0;
    for (int e = 0; e < kE; ++e) { offsets[e] = s; cursors[e] = s; s += counts[e]; }
  }
}

__global__ __launch_bounds__(256) void moe_scatter(const int* __restrict__ topk_i,
                                                   const float* __restrict__ topk_w,
                                                   int* __restrict__ cursors,
                                                   int* __restrict__ tlist,
                                                   float* __restrict__ twt) {
  int t = blockIdx.x * blockDim.x + threadIdx.x;
  if (t >= kT) return;
#pragma unroll
  for (int k = 0; k < kTopK; ++k) {
    int e = topk_i[t * 2 + k];
    int pos = atomicAdd(&cursors[e], 1);
    tlist[pos] = t;
    twt[pos]  = topk_w[t * 2 + k];
  }
}

// ---------------------------------------------------------------------------
// Grouped SwiGLU expert GEMM, double-buffered pipeline:
//   X panel   : per-lane global_load_async_to_lds_b64 (gathered rows, ASYNCcnt)
//   W panels  : ONE tensor_load_to_lds per panel, issued by wave 0 (TENSORcnt)
// Pipeline per step s: wait asynccnt/tensorcnt; barrier; issue(s+1, other buf);
//                      wmma(step s)  <-- overlapped with DMA for s+1.
// Weights pre-transposed: wgT/wuT = [E][F][D], wdT = [E][D][F] (contiguous in K).
// ---------------------------------------------------------------------------
__global__ __launch_bounds__(256) void moe_expert_gemm(
    const unsigned short* __restrict__ xb,
    const unsigned short* __restrict__ wgT,
    const unsigned short* __restrict__ wuT,
    const unsigned short* __restrict__ wdT,
    const unsigned short* __restrict__ zpad,
    const int* __restrict__ counts, const int* __restrict__ offsets,
    const int* __restrict__ tlist, const float* __restrict__ twt,
    float* __restrict__ out) {
  __shared__ unsigned short Hs[BM][HS_STRIDE];      // 33280 B
  __shared__ unsigned short Xs[2][BM][BK + PAD];    //  5120 B (A panel, double buffered)
  __shared__ unsigned short Wa[2][BN][BK + PAD];    // 10240 B (B panel [n][k], dbl buf)
  __shared__ unsigned short Wb[2][BN][BK + PAD];    // 10240 B
  __shared__ int   ttok[BM];
  __shared__ float trw[BM];

  const int e = blockIdx.y;
  const int cnt = counts[e];
  const int rowBase = blockIdx.x * BM;
  if (rowBase >= cnt) return;
  const int off = offsets[e];

  const int tid  = threadIdx.x;
  const int lane = tid & 31;
  const int wave = tid >> 5;
  const int mi = wave & 1;                 // row tile (0..1)
  const int nj = wave >> 1;                // 16-col group (0..3)
  const int mrow = mi * 16 + (lane & 15);  // A-matrix row for this lane
  const int kbA  = (lane >> 4) * 16;       // A layout: lanes 16-31 hold k=16..31
  const int kbB  = (lane >> 4) * 16;       // B layout: lanes 16-31 hold k=16..31
  const int ncol = nj * 16 + (lane & 15);  // B-panel row (n) for this lane

  if (tid < BM) {
    int r = rowBase + tid;
    if (r < cnt) { ttok[tid] = tlist[off + r]; trw[tid] = twt[off + r]; }
    else         { ttok[tid] = -1;            trw[tid] = 0.f; }
  }
  __syncthreads();

  // X staging roles (all 256 threads uniform)
  const int xm = tid >> 3;                 // 0..31: X row
  const int xk = (tid & 7) * 4;            // 8B chunks cover 32 k's

  const unsigned XB = BM * (BK + PAD) * 2; // 2560B per X buffer
  const unsigned WB = BN * (BK + PAD) * 2; // 5120B per W buffer
  const unsigned ldsX  = lds_off(&Xs[0][0][0]) + (unsigned)(xm * (BK + PAD) + xk) * 2u;
  const unsigned ldsWaB = lds_off(&Wa[0][0][0]);   // TDM tile base (uniform)
  const unsigned ldsWbB = lds_off(&Wb[0][0][0]);

  const int xtok = ttok[xm];
  const bool xvalid = xtok >= 0;
  const unsigned short* xrow = xvalid ? (xb + (size_t)xtok * kD + xk) : zpad;

  const size_t wgBase = (size_t)e * kF * kD;  // transposed [F][D]
  const size_t wdBase = (size_t)e * kD * kF;  // transposed [D][F]

  const v8u g1p1 = tdm_group1(kD, kF, kD);    // phase-1 tensors: [F rows][D cols]
  const v8u g1p2 = tdm_group1(kF, kD, kF);    // phase-2 tensor:  [D rows][F cols]

  // ---------------- Phase 1: H = silu(X*Wg) * (X*Wu) ----------------
  // flat pipelined loop: 8 fc-chunks x 32 K-steps = 256 steps
  auto issue1 = [&](int s) {
    const unsigned p = (unsigned)(s & 1);
    const int fc = (s >> 5) << 6;
    const int k0 = (s & 31) << 5;
    async_b64(ldsX + p * XB, xvalid ? (const void*)(xrow + k0) : (const void*)xrow);
    if (wave == 0) {   // one TDM descriptor per weight panel
      tdm_load_2d(ldsWaB + p * WB, wgT + wgBase + (size_t)fc * kD + k0, g1p1);
      tdm_load_2d(ldsWbB + p * WB, wuT + wgBase + (size_t)fc * kD + k0, g1p1);
    }
  };

  issue1(0);
  v8f accG, accU;
#pragma unroll
  for (int i = 0; i < 8; ++i) { accG[i] = 0.f; accU[i] = 0.f; }

#pragma unroll 2
  for (int s = 0; s < 256; ++s) {
    wait_async0();                                     // X panel arrived
    if (wave == 0) __builtin_amdgcn_s_wait_tensorcnt(0);  // W panels arrived
    __syncthreads();     // all waves: data visible AND compute(s-1) finished
    if (s + 1 < 256) issue1(s + 1);   // DMA for s+1 overlaps compute of s
    const int p = s & 1;

    Frag a, bg, bu;
    a.q[0]  = *(const uint4*)&Xs[p][mrow][kbA];
    a.q[1]  = *(const uint4*)&Xs[p][mrow][kbA + 8];
    bg.q[0] = *(const uint4*)&Wa[p][ncol][kbB];
    bg.q[1] = *(const uint4*)&Wa[p][ncol][kbB + 8];
    bu.q[0] = *(const uint4*)&Wb[p][ncol][kbB];
    bu.q[1] = *(const uint4*)&Wb[p][ncol][kbB + 8];
    accG = __builtin_amdgcn_wmma_f32_16x16x32_bf16(false, a.v, false, bg.v,
                                                   (short)0, accG, false, false);
    accU = __builtin_amdgcn_wmma_f32_16x16x32_bf16(false, a.v, false, bu.v,
                                                   (short)0, accU, false, false);

    if ((s & 31) == 31) {   // fc chunk complete: SwiGLU epilogue -> Hs
      const int nc = ((s >> 5) << 6) + ncol;
#pragma unroll
      for (int r = 0; r < 8; ++r) {
        int m = mi * 16 + (lane >> 4) * 8 + r;
        float g = accG[r], u = accU[r];
        float h = (g / (1.f + __expf(-g))) * u;   // silu(g) * u
        Hs[m][nc] = f2bf(h);
      }
#pragma unroll
      for (int i = 0; i < 8; ++i) { accG[i] = 0.f; accU[i] = 0.f; }
    }
  }
  __syncthreads();   // Hs complete; all phase-1 staging reads done

  // ---------------- Phase 2: O = H * Wd, weighted scatter-add ----------------
  // flat pipelined loop: 16 dc-chunks x 16 K-steps = 256 steps
  auto issue2 = [&](int s) {
    const unsigned p = (unsigned)(s & 1);
    const int dc = (s >> 4) << 6;
    const int k0 = (s & 15) << 5;
    if (wave == 0)
      tdm_load_2d(ldsWaB + p * WB, wdT + wdBase + (size_t)dc * kF + k0, g1p2);
  };

  issue2(0);
  v8f accO;
#pragma unroll
  for (int i = 0; i < 8; ++i) accO[i] = 0.f;

#pragma unroll 2
  for (int s = 0; s < 256; ++s) {
    if (wave == 0) __builtin_amdgcn_s_wait_tensorcnt(0);
    __syncthreads();
    if (s + 1 < 256) issue2(s + 1);
    const int p = s & 1;
    const int k0 = (s & 15) << 5;

    Frag a, b;
    a.q[0] = *(const uint4*)&Hs[mrow][k0 + kbA];
    a.q[1] = *(const uint4*)&Hs[mrow][k0 + kbA + 8];
    b.q[0] = *(const uint4*)&Wa[p][ncol][kbB];
    b.q[1] = *(const uint4*)&Wa[p][ncol][kbB + 8];
    accO = __builtin_amdgcn_wmma_f32_16x16x32_bf16(false, a.v, false, b.v,
                                                   (short)0, accO, false, false);

    if ((s & 15) == 15) {   // dc chunk complete: weighted atomic scatter
      const int n = ((s >> 4) << 6) + ncol;
#pragma unroll
      for (int r = 0; r < 8; ++r) {
        int m = mi * 16 + (lane >> 4) * 8 + r;
        int tok = ttok[m];
        if (tok >= 0) atomicAdd(out + (size_t)tok * kD + n, accO[r] * trw[m]);
      }
#pragma unroll
      for (int i = 0; i < 8; ++i) accO[i] = 0.f;
    }
  }
}

// ---------------------------------------------------------------------------
// Host launcher
// ---------------------------------------------------------------------------
extern "C" void kernel_launch(void* const* d_in, const int* in_sizes, int n_in,
                              void* d_out, int out_size, void* d_ws, size_t ws_size,
                              hipStream_t stream) {
  (void)in_sizes; (void)n_in; (void)out_size; (void)ws_size;
  const float* x  = (const float*)d_in[0];   // [T, D]
  const float* rw = (const float*)d_in[1];   // [E, D]
  const float* wg = (const float*)d_in[2];   // [E, D, F]
  const float* wu = (const float*)d_in[3];   // [E, D, F]
  const float* wd = (const float*)d_in[4];   // [E, F, D]
  float* out = (float*)d_out;                // [T, D]

  char* ws = (char*)d_ws;
  size_t o = 0;
  unsigned short* xb  = (unsigned short*)(ws + o); o += (size_t)kT * kD * 2;       // 16 MB
  unsigned short* wgT = (unsigned short*)(ws + o); o += (size_t)kE * kD * kF * 2;  //  8 MB [E][F][D]
  unsigned short* wuT = (unsigned short*)(ws + o); o += (size_t)kE * kD * kF * 2;  //  8 MB [E][F][D]
  unsigned short* wdT = (unsigned short*)(ws + o); o += (size_t)kE * kF * kD * 2;  //  8 MB [E][D][F]
  int*   topk_i = (int*)(ws + o);   o += (size_t)kT * 2 * 4;
  float* topk_w = (float*)(ws + o); o += (size_t)kT * 2 * 4;
  int*   tlist  = (int*)(ws + o);   o += (size_t)kT * 2 * 4;
  float* twt    = (float*)(ws + o); o += (size_t)kT * 2 * 4;
  int*   counts  = (int*)(ws + o);  o += kE * 4;
  int*   offsets = (int*)(ws + o);  o += kE * 4;
  int*   cursors = (int*)(ws + o);  o += kE * 4;
  o = (o + 255) & ~(size_t)255;
  unsigned short* zpad = (unsigned short*)(ws + o); o += 256;  // zero pad for ragged tiles

  hipMemsetAsync(out, 0, (size_t)kT * kD * sizeof(float), stream);
  hipMemsetAsync(counts, 0, kE * sizeof(int), stream);
  hipMemsetAsync(zpad, 0, 256, stream);

  // x: plain bf16 convert (A operand stays [T][D], contiguous in K)
  moe_cvt_bf16<<<4096, 256, 0, stream>>>(x, xb, kT * kD);
  // weights: transpose-convert so B operands are contiguous along K
  {
    dim3 blk(32, 8);
    dim3 g1(kF / 32, kD / 32, kE);   // [E][D][F] -> [E][F][D]
    moe_cvt_t_bf16<<<g1, blk, 0, stream>>>(wg, wgT, kD, kF);
    moe_cvt_t_bf16<<<g1, blk, 0, stream>>>(wu, wuT, kD, kF);
    dim3 g2(kD / 32, kF / 32, kE);   // [E][F][D] -> [E][D][F]
    moe_cvt_t_bf16<<<g2, blk, 0, stream>>>(wd, wdT, kF, kD);
  }

  moe_router<<<kT / 8, 256, 0, stream>>>(x, rw, topk_i, topk_w, counts);
  moe_scan<<<1, 32, 0, stream>>>(counts, offsets, cursors);
  moe_scatter<<<kT / 256, 256, 0, stream>>>(topk_i, topk_w, cursors, tlist, twt);

  dim3 grid((kTopK * kT) / BM, kE);   // 512 tiles x 8 experts; blocks past count exit early
  moe_expert_gemm<<<grid, 256, 0, stream>>>(xb, wgT, wuT, wdT, zpad,
                                            counts, offsets, tlist, twt, out);
}
